// BlockSparseAttention_56573309223858
// MI455X (gfx1250) — matrix operand order repaired and checked by reading, not compiled, not run on hardware
//
#include <hip/hip_runtime.h>
#include <math.h>

typedef __attribute__((ext_vector_type(16))) __bf16 v16bf;
typedef __attribute__((ext_vector_type(8)))  __bf16 bf16x8;
typedef __attribute__((ext_vector_type(4)))  __bf16 bf16x4;
typedef __attribute__((ext_vector_type(8)))  float  v8f;
typedef __attribute__((ext_vector_type(4)))  float  f32x4;

#define S_LEN   4096
#define D_DIM   128
#define BQ      64
#define NBLK    (S_LEN / BQ)     // 64 query/key blocks
#define NWAVES  4
#define NTHREADS (NWAVES * 32)

// ---- WMMA fragment loaders from LDS (bf16) -------------------------------
// A-matrix 16x32 bf16 layout (ISA 7.12.2): lanes 0-15 row m=lane, elems 0..7
// hold K = 8*hi..8*hi+7 (hi = lane>=16), elems 8..15 hold K = 16+8*hi.. .
__device__ __forceinline__ v16bf load_frag_A(const __bf16* base, int row,
                                             int rowStride, int chunk, int hi) {
  const __bf16* p = base + row * rowStride + chunk * 32 + hi * 8;
  bf16x8 lo = *(const bf16x8*)(p);
  bf16x8 hb = *(const bf16x8*)(p + 16);
  return __builtin_shufflevector(lo, hb, 0,1,2,3,4,5,6,7,8,9,10,11,12,13,14,15);
}

// B-matrix 32x16 bf16: lane n = column; elems 0..7 hold K = 8*hi..8*hi+7,
// elems 8..15 hold K = 16+8*hi.. (same K order as A).
__device__ __forceinline__ v16bf load_frag_B(const __bf16* base, int col,
                                             int rowStride, int chunk, int hi) {
  const __bf16* p = base + col * rowStride + chunk * 32 + hi * 8;
  bf16x8 lo = *(const bf16x8*)(p);
  bf16x8 hb = *(const bf16x8*)(p + 16);
  return __builtin_shufflevector(lo, hb, 0,1,2,3,4,5,6,7,8,9,10,11,12,13,14,15);
}

__global__ __launch_bounds__(NTHREADS)
void bsattn_wmma_kernel(const float* __restrict__ q,
                        const float* __restrict__ k,
                        const float* __restrict__ v,
                        float* __restrict__ out) {
  __shared__ __bf16 Qs[BQ * D_DIM];          // 16 KB
  __shared__ __bf16 Ks[BQ * D_DIM];          // 16 KB  (row-major [key][d])
  __shared__ __bf16 Vt[D_DIM * BQ];          // 16 KB  (transposed [d][key])
  __shared__ __bf16 Pw[NWAVES * 16 * BQ];    //  8 KB  (per-wave probs patch)

  const int qb   = blockIdx.x;               // query block 0..63
  const int h    = blockIdx.y;               // head 0..15
  const int tid  = threadIdx.x;
  const int wave = tid >> 5;
  const int lane = tid & 31;
  const int hl   = lane & 15;                // half-lane
  const int hi   = lane >> 4;                // 0/1

  // 1/sqrt(128) * log2(e): softmax evaluated in exp2 domain
  const float sc2 = 0.08838834764831845f * 1.4426950408889634f;

  const float* qg = q + ((size_t)h * S_LEN + (size_t)qb * BQ) * D_DIM;

  // ---- stage Q block (64x128 f32 -> bf16, coalesced float4 reads) --------
  #pragma unroll
  for (int i = 0; i < 16; ++i) {
    int idx = tid + NTHREADS * i;            // 0..2047 float4s
    int r   = idx >> 5;                      // row 0..63
    int c4  = idx & 31;                      // float4 within row
    f32x4 d = *(const f32x4*)(qg + r * D_DIM + c4 * 4);
    bf16x4 b; b[0] = (__bf16)d[0]; b[1] = (__bf16)d[1];
              b[2] = (__bf16)d[2]; b[3] = (__bf16)d[3];
    *(bf16x4*)(Qs + r * D_DIM + c4 * 4) = b;
  }
  __syncthreads();

  // ---- each wave caches its 16 query rows as A-fragments -----------------
  v16bf qa[4];
  #pragma unroll
  for (int c = 0; c < 4; ++c)
    qa[c] = load_frag_A(Qs + wave * 16 * D_DIM, hl, D_DIM, c, hi);

  // online-softmax state; C-layout: v8f elem r, row = 8*hi + r, col = hl
  float mrow[8], lrow[8];
  v8f   o[8];
  #pragma unroll
  for (int r = 0; r < 8; ++r) { mrow[r] = -INFINITY; lrow[r] = 0.0f; }
  #pragma unroll
  for (int f = 0; f < 8; ++f) o[f] = (v8f)(0.0f);

  const int kb0 = (qb > 0) ? qb - 1 : 0;
  const int kb1 = (qb < NBLK - 1) ? qb + 1 : NBLK - 1;

  for (int kb = kb0; kb <= kb1; ++kb) {
    __syncthreads();   // previous Ks/Vt fully consumed
    const float* kg = k + ((size_t)h * S_LEN + (size_t)kb * BQ) * D_DIM;
    const float* vg = v + ((size_t)h * S_LEN + (size_t)kb * BQ) * D_DIM;
    #pragma unroll
    for (int i = 0; i < 16; ++i) {
      int idx = tid + NTHREADS * i;
      int r   = idx >> 5;
      int c4  = idx & 31;
      f32x4 dk = *(const f32x4*)(kg + r * D_DIM + c4 * 4);
      bf16x4 bk; bk[0] = (__bf16)dk[0]; bk[1] = (__bf16)dk[1];
                 bk[2] = (__bf16)dk[2]; bk[3] = (__bf16)dk[3];
      *(bf16x4*)(Ks + r * D_DIM + c4 * 4) = bk;
      f32x4 dv = *(const f32x4*)(vg + r * D_DIM + c4 * 4);
      #pragma unroll
      for (int j = 0; j < 4; ++j)
        Vt[(c4 * 4 + j) * BQ + r] = (__bf16)dv[j];
    }
    __syncthreads();

    // ---- prefetch next key block's K/V while we compute on this one ------
    if (kb < kb1) {
      const char* kn = (const char*)(kg + (size_t)BQ * D_DIM);
      const char* vn = (const char*)(vg + (size_t)BQ * D_DIM);
      __builtin_prefetch(kn + tid * 256, 0, 3);
      __builtin_prefetch(kn + tid * 256 + 128, 0, 3);
      __builtin_prefetch(vn + tid * 256, 0, 3);
      __builtin_prefetch(vn + tid * 256 + 128, 0, 3);
    }

    // ---- S = Q K^T : 4 key subtiles x 4 K-chunks = 16 WMMAs --------------
    v8f s[4];
    #pragma unroll
    for (int n = 0; n < 4; ++n) {
      v16bf kf[4];
      #pragma unroll
      for (int c = 0; c < 4; ++c)          // batch: 8x ds_load_b128, one wait
        kf[c] = load_frag_B(Ks, n * 16 + hl, D_DIM, c, hi);
      v8f acc = (v8f)(0.0f);
      #pragma unroll
      for (int c = 0; c < 4; ++c)
        acc = __builtin_amdgcn_wmma_f32_16x16x32_bf16(
                  false, qa[c], false, kf[c], (short)0, acc, false, false);
      #pragma unroll
      for (int r = 0; r < 8; ++r) acc[r] *= sc2;   // exp2-domain scores
      s[n] = acc;
    }

    // ---- online softmax (exp2 domain) ------------------------------------
    float rmax[8];
    #pragma unroll
    for (int r = 0; r < 8; ++r) {
      float m0 = fmaxf(fmaxf(s[0][r], s[1][r]), fmaxf(s[2][r], s[3][r]));
      #pragma unroll
      for (int msk = 1; msk < 16; msk <<= 1)
        m0 = fmaxf(m0, __shfl_xor(m0, msk, 32));
      rmax[r] = m0;
    }
    float alpha[8], rsum[8];
    #pragma unroll
    for (int r = 0; r < 8; ++r) {
      float mn = fmaxf(mrow[r], rmax[r]);
      alpha[r] = exp2f(mrow[r] - mn);
      mrow[r]  = mn;
      rsum[r]  = 0.0f;
    }
    __bf16* pw = Pw + wave * 16 * BQ;
    #pragma unroll
    for (int n = 0; n < 4; ++n)
      #pragma unroll
      for (int r = 0; r < 8; ++r) {
        float p = exp2f(s[n][r] - mrow[r]);
        rsum[r] += p;
        pw[(hi * 8 + r) * BQ + n * 16 + hl] = (__bf16)p;
      }
    #pragma unroll
    for (int r = 0; r < 8; ++r) {
      float t = rsum[r];
      #pragma unroll
      for (int msk = 1; msk < 16; msk <<= 1)
        t += __shfl_xor(t, msk, 32);
      lrow[r] = lrow[r] * alpha[r] + t;
    }

    // ---- O = O*alpha + P V : 8 feat subtiles x 2 key chunks = 16 WMMAs ---
    v16bf pa[2];
    #pragma unroll
    for (int c = 0; c < 2; ++c)
      pa[c] = load_frag_A(pw, hl, BQ, c, hi);
    #pragma unroll
    for (int f = 0; f < 8; ++f) {
      v16bf vf[2];
      #pragma unroll
      for (int c = 0; c < 2; ++c)          // batch: 4x ds_load_b128, one wait
        vf[c] = load_frag_B(Vt, f * 16 + hl, BQ, c, hi);
      v8f acc = o[f];
      #pragma unroll
      for (int r = 0; r < 8; ++r) acc[r] *= alpha[r];
      #pragma unroll
      for (int c = 0; c < 2; ++c)
        acc = __builtin_amdgcn_wmma_f32_16x16x32_bf16(
                  false, pa[c], false, vf[c], (short)0, acc, false, false);
      o[f] = acc;
    }
  }

  // ---- normalize and store ------------------------------------------------
  float inv[8];
  #pragma unroll
  for (int r = 0; r < 8; ++r) inv[r] = 1.0f / lrow[r];

  const int rowbase = qb * BQ + wave * 16 + hi * 8;
  #pragma unroll
  for (int f = 0; f < 8; ++f)
    #pragma unroll
    for (int r = 0; r < 8; ++r) {
      size_t gi = ((size_t)h * S_LEN + rowbase + r) * D_DIM + f * 16 + hl;
      out[gi] = o[f][r] * inv[r];
    }
}

extern "C" void kernel_launch(void* const* d_in, const int* in_sizes, int n_in,
                              void* d_out, int out_size, void* d_ws, size_t ws_size,
                              hipStream_t stream) {
  const float* q = (const float*)d_in[0];
  const float* k = (const float*)d_in[1];
  const float* v = (const float*)d_in[2];
  float* out = (float*)d_out;
  dim3 grid(NBLK, 16);          // (query blocks, heads)
  dim3 block(NTHREADS);
  bsattn_wmma_kernel<<<grid, block, 0, stream>>>(q, k, v, out);
}
